// LearnedAlgorithm_29996051595788
// MI455X (gfx1250) — matrix operand order repaired and checked by reading, not compiled, not run on hardware
//
#include <hip/hip_runtime.h>
#include <hip/hip_bf16.h>

#define EPSV 1e-10f

typedef __attribute__((ext_vector_type(16))) __bf16 v16bf;
typedef __attribute__((ext_vector_type(8)))  float  v8f;

#define WMMA_BF16(A, B, C) \
  __builtin_amdgcn_wmma_f32_16x16x32_bf16(false, (A), false, (B), (short)0, (C), false, false)

// ---------------------------------------------------------------- reductions
__device__ __forceinline__ float wave_sum(float v) {
#pragma unroll
  for (int o = 16; o > 0; o >>= 1) v += __shfl_xor(v, o, 32);
  return v;
}
__device__ __forceinline__ float wave_max(float v) {
#pragma unroll
  for (int o = 16; o > 0; o >>= 1) v = fmaxf(v, __shfl_xor(v, o, 32));
  return v;
}

__global__ void init_acc(float* acc) {
  if (threadIdx.x < 4) acc[threadIdx.x] = 0.0f;  // 0.0f bits == uint 0 (max identity)
}

__global__ void __launch_bounds__(256) reduce4(const float* __restrict__ grad,
                                               const float* __restrict__ s0,
                                               const float* __restrict__ s1,
                                               float* __restrict__ acc, int n4) {
  float g2 = 0.f, d2 = 0.f, gd = 0.f, mg = 0.f;
  int stride = gridDim.x * blockDim.x;
  for (int i = blockIdx.x * blockDim.x + threadIdx.x; i < n4; i += stride) {
    float4 g = ((const float4*)grad)[i];
    float4 a = ((const float4*)s0)[i];
    float4 b = ((const float4*)s1)[i];
    float d;
    d = b.x - a.x; g2 += g.x * g.x; d2 += d * d; gd += g.x * d; mg = fmaxf(mg, fabsf(g.x));
    d = b.y - a.y; g2 += g.y * g.y; d2 += d * d; gd += g.y * d; mg = fmaxf(mg, fabsf(g.y));
    d = b.z - a.z; g2 += g.z * g.z; d2 += d * d; gd += g.z * d; mg = fmaxf(mg, fabsf(g.z));
    d = b.w - a.w; g2 += g.w * g.w; d2 += d * d; gd += g.w * d; mg = fmaxf(mg, fabsf(g.w));
  }
  g2 = wave_sum(g2); d2 = wave_sum(d2); gd = wave_sum(gd); mg = wave_max(mg);
  __shared__ float sm[4][8];
  int lane = threadIdx.x & 31, w = threadIdx.x >> 5;
  if (lane == 0) { sm[0][w] = g2; sm[1][w] = d2; sm[2][w] = gd; sm[3][w] = mg; }
  __syncthreads();
  if (threadIdx.x == 0) {
    float t0 = 0.f, t1 = 0.f, t2 = 0.f, t3 = 0.f;
#pragma unroll
    for (int i = 0; i < 8; ++i) {
      t0 += sm[0][i]; t1 += sm[1][i]; t2 += sm[2][i]; t3 = fmaxf(t3, sm[3][i]);
    }
    atomicAdd(&acc[0], t0);
    atomicAdd(&acc[1], t1);
    atomicAdd(&acc[2], t2);
    atomicMax((unsigned int*)&acc[3], __float_as_uint(t3));  // t3 >= 0: uint order == float order
  }
}

// ----------------------------------------------------- scalar coefficient MLP
__global__ void coeff_mlp(const float* __restrict__ acc,
                          const float* __restrict__ loss_cur, const float* __restrict__ loss_old,
                          const int* __restrict__ iter,
                          const float* __restrict__ lw0, const float* __restrict__ lb0,
                          const float* __restrict__ lw1, const float* __restrict__ lb1,
                          const float* __restrict__ lw2, const float* __restrict__ lb2,
                          const float* __restrict__ lw3, const float* __restrict__ lb3,
                          float* __restrict__ scal) {
  if (threadIdx.x != 0 || blockIdx.x != 0) return;
  float gn = sqrtf(acc[0]);
  float dn = sqrtf(acc[1]);
  float gd = acc[2];
  float mg = acc[3];  // stored as nonneg-float bits
  float ig  = (gn > EPSV) ? (1.0f / gn) : 1.0f;
  float idn = (dn > EPSV) ? (1.0f / dn) : 1.0f;
  float it = (float)iter[0];
  float f[6] = { log1pf(gn), log1pf(dn), gd * ig * idn, mg * ig, it,
                 logf(loss_cur[0]) - logf(loss_old[0]) };
  float h0[30], h1[20], h2[10], c[4];
  for (int o = 0; o < 30; ++o) { float s = lb0[o]; for (int i = 0; i < 6;  ++i) s += lw0[o * 6  + i] * f[i];  h0[o] = fmaxf(s, 0.f); }
  for (int o = 0; o < 20; ++o) { float s = lb1[o]; for (int i = 0; i < 30; ++i) s += lw1[o * 30 + i] * h0[i]; h1[o] = fmaxf(s, 0.f); }
  for (int o = 0; o < 10; ++o) { float s = lb2[o]; for (int i = 0; i < 20; ++i) s += lw2[o * 20 + i] * h1[i]; h2[o] = fmaxf(s, 0.f); }
  for (int o = 0; o < 4;  ++o) { float s = lb3[o]; for (int i = 0; i < 10; ++i) s += lw3[o * 10 + i] * h2[i]; c[o]  = s; }
  scal[0] = c[0] * ig;   // g-channel 0 (with gradient_param)
  scal[1] = c[1] * idn;  // d-channel 1 (with extrapolation)
  scal[2] = c[2] * ig;   // g-channel 2
  scal[3] = c[3] * idn;  // d-channel 3
  scal[4] = rsqrtf(1.0f + it);
}

// --------------------------------------------------------- WMMA fused update
//
// K layout: D-fragments put h rows {0-7,16-23} in lanes 0-15 and
// {8-15,24-31} in lanes 16-31; B-fragments (table order) take K {0-7,16-23} in lanes 0-15 and
// K {8-15,24-31} in lanes 16-31, same as D rows. Table-order K (a_kslot)
// in the *weight* (A) loads of layers 1-3 makes the D->B repack fully
// lane-local: no ds_bpermute at all.
__device__ __forceinline__ int sigma_k(int k) {
  if (k >= 8 && k < 16) return k + 8;   // K 8-15  <- h rows 16-23
  if (k >= 16 && k < 24) return k - 8;  // K 16-23 <- h rows 8-15
  return k;
}

// A-fragment K slot for (lane, j) per ISA 7.12.2 (16-bit A 16x32).
__device__ __forceinline__ int a_kslot(int j, int lane) {
  return j + ((j >= 8) ? 8 : 0) + ((lane & 16) ? 8 : 0);
}

// Layers 1-3: K columns in table order (a_kslot). m_dup: if >=0, weight row 0 is
// replicated at tile rows 0 and m_dup (used to mirror layer-3 output into the
// hi lane half so every lane stores its own element).
__device__ __forceinline__ v16bf load_A_perm(const float* W, int rows, int cols,
                                             int mbase, int lane, int m_dup) {
  v16bf a;
  int m = mbase + (lane & 15);
  if (m_dup >= 0) m = (m == 0 || m == m_dup) ? 0 : rows;  // rows => padded zero
#pragma unroll
  for (int j = 0; j < 16; ++j) {
    int c = a_kslot(j, lane);
    float w = (m < rows && c < cols) ? W[m * cols + c] : 0.0f;
    a[j] = (__bf16)w;
  }
  return a;
}

// Layer 0: natural K; channel columns placed at K base 0 (tile 0) or 8
// (tile 1, so hi lanes feed their local element into B with no shuffle).
__device__ __forceinline__ v16bf load_A0(const float* W, int rows, int mbase,
                                         int lane, int kbase) {
  v16bf a;
  int m = mbase + (lane & 15);
#pragma unroll
  for (int j = 0; j < 16; ++j) {
    int c = a_kslot(j, lane) - kbase;
    float w = (m < rows && c >= 0 && c < 4) ? W[m * 4 + c] : 0.0f;
    a[j] = (__bf16)w;
  }
  return a;
}

// Bias + ReLU + repack: fully lane-local thanks to the table-order K layout.
__device__ __forceinline__ v16bf act_pack(v8f Da, v8f Db, const float* bA, const float* bB) {
  v16bf b;
#pragma unroll
  for (int j = 0; j < 8; ++j) b[j]     = (__bf16)fmaxf(Da[j] + bA[j], 0.0f);
#pragma unroll
  for (int j = 0; j < 8; ++j) b[j + 8] = (__bf16)fmaxf(Db[j] + bB[j], 0.0f);
  return b;
}

__global__ void __launch_bounds__(256) fused_update(
    const float* __restrict__ grad, const float* __restrict__ s0, const float* __restrict__ s1,
    const float* __restrict__ gp, const float* __restrict__ ex,
    const float* __restrict__ cw0, const float* __restrict__ cb0,
    const float* __restrict__ cw1, const float* __restrict__ cb1,
    const float* __restrict__ cw2, const float* __restrict__ cb2,
    const float* __restrict__ cw3, const float* __restrict__ cb3,
    const float* __restrict__ scal, float* __restrict__ out, int n) {
  const int lane = threadIdx.x & 31;
  const int wave = threadIdx.x >> 5;
  const bool hi = (lane & 16) != 0;

  // Weight fragments (zero-padded; layer-0 in two K-base variants).
  v16bf A0aL = load_A0(cw0, 20, 0,  lane, 0);
  v16bf A0bL = load_A0(cw0, 20, 16, lane, 0);
  v16bf A0aH = load_A0(cw0, 20, 0,  lane, 8);
  v16bf A0bH = load_A0(cw0, 20, 16, lane, 8);
  v16bf A1a  = load_A_perm(cw1, 20, 20, 0,  lane, -1);
  v16bf A1b  = load_A_perm(cw1, 20, 20, 16, lane, -1);
  v16bf A2a  = load_A_perm(cw2, 20, 20, 0,  lane, -1);
  v16bf A2b  = load_A_perm(cw2, 20, 20, 16, lane, -1);
  v16bf A3   = load_A_perm(cw3, 1,  20, 0,  lane, 8);  // output row mirrored at M=0 and M=8

  float b0A[8], b0B[8], b1A[8], b1B[8], b2A[8], b2B[8];
#pragma unroll
  for (int r = 0; r < 8; ++r) {
    int ra = r + (hi ? 8 : 0);  // D-tile-a row this lane holds
    int rb = ra + 16;           // D-tile-b absolute row
    b0A[r] = cb0[ra]; b0B[r] = (rb < 20) ? cb0[rb] : 0.0f;
    b1A[r] = cb1[ra]; b1B[r] = (rb < 20) ? cb1[rb] : 0.0f;
    b2A[r] = cb2[ra]; b2B[r] = (rb < 20) ? cb2[rb] : 0.0f;
  }
  const float a0 = scal[0], a1 = scal[1], a2 = scal[2], a3 = scal[3], osc = scal[4];
  const float cb3v = cb3[0];
  const v8f zero = {0.f, 0.f, 0.f, 0.f, 0.f, 0.f, 0.f, 0.f};

  // Each wave: 32 elements per iteration, two independent 16-column N tiles.
  for (int base = blockIdx.x * 256 + wave * 32; base < n; base += gridDim.x * 256) {
    int e = base + lane;
    float gr  = grad[e];
    float df  = s1[e] - s0[e];
    float gv  = gp[e];
    float ev  = ex[e];
    float s1v = s1[e];
    float x0 = a0 * gv * gr, x1 = a1 * ev * df, x2 = a2 * gr, x3 = a3 * df;

    // Layer-0 B fragments: tile 0 live in lo lanes (K0-3), tile 1 live in hi
    // lanes (K8-11) — each lane contributes only its own element. No shuffles.
    v16bf B0, B1;
#pragma unroll
    for (int j = 0; j < 16; ++j) { B0[j] = (__bf16)0.0f; B1[j] = (__bf16)0.0f; }
    B0[0] = (__bf16)(hi ? 0.0f : x0);
    B0[1] = (__bf16)(hi ? 0.0f : x1);
    B0[2] = (__bf16)(hi ? 0.0f : x2);
    B0[3] = (__bf16)(hi ? 0.0f : x3);
    B1[0] = (__bf16)(hi ? x0 : 0.0f);
    B1[1] = (__bf16)(hi ? x1 : 0.0f);
    B1[2] = (__bf16)(hi ? x2 : 0.0f);
    B1[3] = (__bf16)(hi ? x3 : 0.0f);

    // Two independent WMMA chains, interleaved so act_pack of one tile fills
    // the other tile's WMMA hazard slots.
    v8f D0a = WMMA_BF16(A0aL, B0, zero);
    v8f D0b = WMMA_BF16(A0bL, B0, zero);
    v8f E0a = WMMA_BF16(A0aH, B1, zero);
    v8f E0b = WMMA_BF16(A0bH, B1, zero);
    v16bf H1 = act_pack(D0a, D0b, b0A, b0B);
    v16bf G1 = act_pack(E0a, E0b, b0A, b0B);

    v8f D1a = WMMA_BF16(A1a, H1, zero);
    v8f D1b = WMMA_BF16(A1b, H1, zero);
    v8f E1a = WMMA_BF16(A1a, G1, zero);
    v8f E1b = WMMA_BF16(A1b, G1, zero);
    v16bf H2 = act_pack(D1a, D1b, b1A, b1B);
    v16bf G2 = act_pack(E1a, E1b, b1A, b1B);

    v8f D2a = WMMA_BF16(A2a, H2, zero);
    v8f D2b = WMMA_BF16(A2b, H2, zero);
    v8f E2a = WMMA_BF16(A2a, G2, zero);
    v8f E2b = WMMA_BF16(A2b, G2, zero);
    v16bf H3 = act_pack(D2a, D2b, b2A, b2B);
    v16bf G3 = act_pack(E2a, E2b, b2A, b2B);

    v8f D3 = WMMA_BF16(A3, H3, zero);  // row 0 -> lo lanes (tile-0 elements)
    v8f E3 = WMMA_BF16(A3, G3, zero);  // row 8 mirror -> hi lanes (tile-1 elements)

    // Every lane stores its own element: one coalesced 128B store, no shuffle.
    float o = hi ? E3[0] : D3[0];
    out[e] = s1v + (o + cb3v) * osc;
  }
}

// ------------------------------------------------------------------- launch
extern "C" void kernel_launch(void* const* d_in, const int* in_sizes, int n_in,
                              void* d_out, int out_size, void* d_ws, size_t ws_size,
                              hipStream_t stream) {
  const float* grad     = (const float*)d_in[0];
  const float* s0       = (const float*)d_in[1];
  const float* s1       = (const float*)d_in[2];
  const float* loss_cur = (const float*)d_in[3];
  const float* loss_old = (const float*)d_in[4];
  const int*   iter     = (const int*)d_in[5];
  const float* gp       = (const float*)d_in[6];
  const float* ex       = (const float*)d_in[7];
  const float* cw0 = (const float*)d_in[8],  *cb0 = (const float*)d_in[9];
  const float* cw1 = (const float*)d_in[10], *cb1 = (const float*)d_in[11];
  const float* cw2 = (const float*)d_in[12], *cb2 = (const float*)d_in[13];
  const float* cw3 = (const float*)d_in[14], *cb3 = (const float*)d_in[15];
  const float* lw0 = (const float*)d_in[16], *lb0 = (const float*)d_in[17];
  const float* lw1 = (const float*)d_in[18], *lb1 = (const float*)d_in[19];
  const float* lw2 = (const float*)d_in[20], *lb2 = (const float*)d_in[21];
  const float* lw3 = (const float*)d_in[22], *lb3 = (const float*)d_in[23];
  float* out  = (float*)d_out;
  float* acc  = (float*)d_ws;   // [0..3] reductions
  float* scal = acc + 4;        // [4..8] coeff scales + output scale
  int n = in_sizes[0];          // DIM (multiple of 512)

  init_acc<<<1, 32, 0, stream>>>(acc);
  reduce4<<<1024, 256, 0, stream>>>(grad, s0, s1, acc, n / 4);
  coeff_mlp<<<1, 1, 0, stream>>>(acc, loss_cur, loss_old, iter,
                                 lw0, lb0, lw1, lb1, lw2, lb2, lw3, lb3, scal);
  fused_update<<<2048, 256, 0, stream>>>(grad, s0, s1, gp, ex,
                                         cw0, cb0, cw1, cb1, cw2, cb2, cw3, cb3,
                                         scal, out, n);
}
